// SambaReadout_13804024889517
// MI455X (gfx1250) — matrix operand order repaired and checked by reading, not compiled, not run on hardware
//
#include <hip/hip_runtime.h>

// ---------------------------------------------------------------------------
// SambaReadout for MI455X (gfx1250, wave32, WMMA + async global->LDS)
// L=12, B=1, S=1024, SD=16384, QK=128, H=512, V=32000
// ---------------------------------------------------------------------------

typedef __attribute__((ext_vector_type(16))) __bf16          v16bf;
typedef __attribute__((ext_vector_type(16))) unsigned short  v16u;
typedef __attribute__((ext_vector_type(8)))  float           v8f;

#define L_LAYERS 12
#define SEQ      1024
#define SD       16384
#define QK       128
#define HID      512
#define VOCAB    32000
#define NCAT     768            // QK + HID + QK  (k | v | qpart)
#define SCALE_QK 0.08838834764831845f

__device__ __forceinline__ unsigned short f2bf(float f) {
  unsigned int u = __float_as_uint(f);
  u += 0x7FFFu + ((u >> 16) & 1u);          // round-to-nearest-even
  return (unsigned short)(u >> 16);
}
__device__ __forceinline__ unsigned pack2bf(float lo, float hi) {
  return (unsigned)f2bf(lo) | ((unsigned)f2bf(hi) << 16);
}

// --- CDNA5 async global->LDS copy (ASYNCcnt-tracked DMA, no VGPR data) -----
// LDS address = low 32 bits of the generic pointer (ISA 10.2 aperture rule).
__device__ __forceinline__ void async_ld_b128(const void* gptr, void* lptr) {
  unsigned           loff = (unsigned)(unsigned long long)lptr;
  unsigned long long ga   = (unsigned long long)gptr;
  asm volatile("global_load_async_to_lds_b128 %0, %1, off"
               :: "v"(loff), "v"(ga) : "memory");
}
__device__ __forceinline__ void wait_asynccnt0() {
  asm volatile("s_wait_asynccnt 0x0" ::: "memory");
}

// ---------------------------------------------------------------------------
// h (f32) -> bf16, same row-major layout.  8 elements / thread.
// ---------------------------------------------------------------------------
__global__ void f32_to_bf16_kernel(const float* __restrict__ src,
                                   unsigned short* __restrict__ dst,
                                   long long n8) {
  long long i = (long long)blockIdx.x * blockDim.x + threadIdx.x;
  if (i >= n8) return;
  const float4* s = (const float4*)src + i * 2;
  float4 a = s[0], b = s[1];
  uint4 o;
  o.x = pack2bf(a.x, a.y); o.y = pack2bf(a.z, a.w);
  o.z = pack2bf(b.x, b.y); o.w = pack2bf(b.z, b.w);
  ((uint4*)dst)[i] = o;
}

// ---------------------------------------------------------------------------
// Pack a KxN f32 row-major matrix into bf16 WMMA B-fragments (contiguous k order).
// Fragment (nb,kb): 32 lanes x 16 bf16; lane holds col n = nb*16+(lane&15),
// K rows kb*32 + (lane&16 ? 16:0) + i.
// dst index = (((nbOffset+nb)*kbTotal + kb)*32 + lane)*16 + i
// ---------------------------------------------------------------------------
__global__ void pack_frags_kernel(const float* __restrict__ src, int K, int N,
                                  unsigned short* __restrict__ dst,
                                  int nbOffset, int kbTotal) {
  long long idx   = (long long)blockIdx.x * blockDim.x + threadIdx.x;
  long long total = (long long)(N >> 4) * kbTotal * 512;
  if (idx >= total) return;
  int i    = (int)(idx & 15);
  int lane = (int)((idx >> 4) & 31);
  long long t = idx >> 9;
  int kb  = (int)(t % kbTotal);
  int nbL = (int)(t / kbTotal);
  int k = kb * 32 + ((lane & 16) ? 16 : 0) + i;
  int n = nbL * 16 + (lane & 15);
  long long dstIdx = (((long long)(nbOffset + nbL) * kbTotal + kb) * 32 + lane) * 16 + i;
  dst[dstIdx] = f2bf(src[(long long)k * N + n]);
}

// ---------------------------------------------------------------------------
// Tiled GEMM: C[M x Ntot] = A[M x K] * B (bf16, pre-packed frags)
// Block tile 128x128, 256 threads = 8 waves (2M x 4N), each wave 4x2 WMMA
// accumulators of v_wmma_f32_16x16x32_bf16.
// AFP32 == 0: A is bf16 in global; tiles staged with global_load_async_to_lds
//             (ASYNCcnt) into double-buffered LDS (rows padded to 80B).
// AFP32 == 1: A is f32 in global; staged through VGPRs with one bf16
//             conversion per element (fallback when ws can't hold bf16 h).
// MODE 0: epilogue relu(+bias0) cols<128, +bias1 cols<640, raw otherwise.
// MODE 1: epilogue +bias0.
// ---------------------------------------------------------------------------
template <int MODE, int AFP32>
__global__ __launch_bounds__(256) void gemm_wmma_kernel(
    const void* __restrict__ Aopaque, int lda, int K,
    const unsigned short* __restrict__ Bfrag, int kbTotal,
    float* __restrict__ C, int ldc,
    const float* __restrict__ bias0, const float* __restrict__ bias1) {

  __shared__ unsigned short As[2][128][40];   // 80B row stride: 16 distinct banks / b128

  const float*          Af = (const float*)Aopaque;
  const unsigned short* Ab = (const unsigned short*)Aopaque;

  const int tid  = threadIdx.x;
  const int lane = tid & 31;
  const int wave = tid >> 5;
  const int wm   = wave & 1;         // M offset = wm*64
  const int wn   = wave >> 1;        // N offset = wn*32
  const int rowBase = blockIdx.y * 128;
  const int colBase = blockIdx.x * 128;
  const int nbBase  = (colBase >> 4) + wn * 2;

  v8f acc[4][2];
#pragma unroll
  for (int m = 0; m < 4; ++m)
#pragma unroll
    for (int t = 0; t < 2; ++t)
#pragma unroll
      for (int j = 0; j < 8; ++j) acc[m][t][j] = 0.0f;

  // staging coordinates
  const int r    = tid >> 1;                   // f32 path: row staged by this thread
  const int half = tid & 1;
  const int srow0 = wave * 16 + (lane >> 2);   // async path: wave covers 16 rows
  const int qcol  = (lane & 3) * 8;            // 16B quarter of the 32-elem slab

  // ---- prologue: stage tile kb = 0 into buffer 0 ----
  if (AFP32 == 0) {
#pragma unroll
    for (int qq = 0; qq < 2; ++qq) {
      int srow = srow0 + qq * 8;
      async_ld_b128(Ab + (size_t)(rowBase + srow) * lda + qcol,
                    &As[0][srow][qcol]);
    }
    wait_asynccnt0();
  } else {
    const float* gp = Af + (size_t)(rowBase + r) * lda + half * 16;
    float4 x0 = *(const float4*)(gp + 0);
    float4 x1 = *(const float4*)(gp + 4);
    float4 x2 = *(const float4*)(gp + 8);
    float4 x3 = *(const float4*)(gp + 12);
    uint4 o0, o1;
    o0.x = pack2bf(x0.x, x0.y); o0.y = pack2bf(x0.z, x0.w);
    o0.z = pack2bf(x1.x, x1.y); o0.w = pack2bf(x1.z, x1.w);
    o1.x = pack2bf(x2.x, x2.y); o1.y = pack2bf(x2.z, x2.w);
    o1.z = pack2bf(x3.x, x3.y); o1.w = pack2bf(x3.z, x3.w);
    unsigned short* lp = &As[0][r][half * 16];
    *(uint4*)(lp)     = o0;
    *(uint4*)(lp + 8) = o1;
  }
  __syncthreads();

  const int kbase = (lane & 16) ? 16 : 0;  // A-frag K base (elements)
  const int rsel  = lane & 15;

  int buf = 0;
  for (int kb = 0; kb < kbTotal; ++kb) {
    const bool hasNext = (kb + 1 < kbTotal);

    float4 p0, p1, p2, p3;
    if (AFP32 == 0) {
      if (hasNext) {
#pragma unroll
        for (int qq = 0; qq < 2; ++qq) {
          int srow = srow0 + qq * 8;
          async_ld_b128(Ab + (size_t)(rowBase + srow) * lda + (kb + 1) * 32 + qcol,
                        &As[buf ^ 1][srow][qcol]);
        }
      }
    } else {
      if (hasNext) {
        const float* gp = Af + (size_t)(rowBase + r) * lda + (kb + 1) * 32 + half * 16;
        p0 = *(const float4*)(gp + 0);
        p1 = *(const float4*)(gp + 4);
        p2 = *(const float4*)(gp + 8);
        p3 = *(const float4*)(gp + 12);
      }
    }
    if (hasNext)   // L2 hint for next B fragments
      __builtin_prefetch(Bfrag + (((size_t)nbBase * kbTotal + kb + 1) * 32 + lane) * 16, 0, 1);

    // B fragments: one contiguous 32B load per lane (L2-resident)
    v16u bu0 = *(const v16u*)(Bfrag + (((size_t)(nbBase + 0) * kbTotal + kb) * 32 + lane) * 16);
    v16u bu1 = *(const v16u*)(Bfrag + (((size_t)(nbBase + 1) * kbTotal + kb) * 32 + lane) * 16);
    v16bf bf0 = __builtin_bit_cast(v16bf, bu0);
    v16bf bf1 = __builtin_bit_cast(v16bf, bu1);

#pragma unroll
    for (int m = 0; m < 4; ++m) {
      const unsigned short* ar = &As[buf][wm * 64 + m * 16 + rsel][0];
      struct { uint4 lo, hi; } au;
      au.lo = *(const uint4*)(ar + kbase);        // K = kbase .. kbase+7
      au.hi = *(const uint4*)(ar + kbase + 8);    // K = kbase+8 .. kbase+15
      v16bf a = __builtin_bit_cast(v16bf, au);

      acc[m][0] = __builtin_amdgcn_wmma_f32_16x16x32_bf16(
          false, a, false, bf0, (short)0, acc[m][0], false, false);
      acc[m][1] = __builtin_amdgcn_wmma_f32_16x16x32_bf16(
          false, a, false, bf1, (short)0, acc[m][1], false, false);
    }

    if (AFP32 == 0) {
      if (hasNext) wait_asynccnt0();
    } else {
      if (hasNext) {
        uint4 o0, o1;
        o0.x = pack2bf(p0.x, p0.y); o0.y = pack2bf(p0.z, p0.w);
        o0.z = pack2bf(p1.x, p1.y); o0.w = pack2bf(p1.z, p1.w);
        o1.x = pack2bf(p2.x, p2.y); o1.y = pack2bf(p2.z, p2.w);
        o1.z = pack2bf(p3.x, p3.y); o1.w = pack2bf(p3.z, p3.w);
        unsigned short* lp = &As[buf ^ 1][r][half * 16];
        *(uint4*)(lp)     = o0;
        *(uint4*)(lp + 8) = o1;
      }
    }
    __syncthreads();
    buf ^= 1;
  }

  // ---- epilogue ----
  // C/D layout: element i -> row = i + (lane&16 ? 8:0), col = lane&15.
  // Region selection and bias value are invariant across the 8 accumulator
  // rows and (for the region) uniform across lanes of a 16-wide N-subtile:
  // hoist them out of the element loop (one scalar branch + one load per t).
  const int rowHi = (lane & 16) ? 8 : 0;
  const int colIn = lane & 15;
#pragma unroll
  for (int t = 0; t < 2; ++t) {
    const int cbase = colBase + wn * 32 + t * 16;   // uniform per subtile
    const int col   = cbase + colIn;
    float badd;
    bool  dorelu = false;
    if (MODE == 0) {
      if (cbase < QK)            { badd = bias0[col];      dorelu = true; }   // k
      else if (cbase < QK + HID) { badd = bias1[col - QK]; }                  // v
      else                       { badd = 0.0f; }                             // q-part
    } else {
      badd = bias0[col];                                                      // b2
    }
#pragma unroll
    for (int m = 0; m < 4; ++m) {
      const int rbase = rowBase + wm * 64 + m * 16 + rowHi;
#pragma unroll
      for (int i = 0; i < 8; ++i) {
        float v = acc[m][t][i] + badd;
        if (dorelu) v = fmaxf(v, 0.0f);
        C[(size_t)(rbase + i) * ldc + col] = v;
      }
    }
  }
}

// ---------------------------------------------------------------------------
// q[s,j] = relu( mean_l qpart[l,s,j] + bq[j] )   (mean commutes with Wq)
// ---------------------------------------------------------------------------
__global__ void compute_q_kernel(const float* __restrict__ kqv,
                                 const float* __restrict__ bq,
                                 float* __restrict__ q) {
  int idx = blockIdx.x * blockDim.x + threadIdx.x;
  if (idx >= SEQ * QK) return;
  int s = idx >> 7, j = idx & 127;
  float acc = 0.0f;
#pragma unroll
  for (int l = 0; l < L_LAYERS; ++l)
    acc += kqv[((size_t)(l * SEQ + s)) * NCAT + QK + HID + j];
  q[idx] = fmaxf(acc * (1.0f / L_LAYERS) + bq[j], 0.0f);
}

// ---------------------------------------------------------------------------
// Per-token: scores over L, softmax, agg = sum_l attn[l]*v[l].  One wave/token.
// ---------------------------------------------------------------------------
__global__ void attn_agg_kernel(const float* __restrict__ kqv,
                                const float* __restrict__ q,
                                float* __restrict__ agg,
                                float* __restrict__ attn_last) {
  const int s    = blockIdx.x;
  const int lane = threadIdx.x;   // 32 lanes (wave32)

  float qv[4];
#pragma unroll
  for (int j = 0; j < 4; ++j) qv[j] = q[s * QK + j * 32 + lane];

  float scores[L_LAYERS];
#pragma unroll
  for (int l = 0; l < L_LAYERS; ++l) {
    const float* krow = kqv + ((size_t)(l * SEQ + s)) * NCAT;
    float p = 0.0f;
#pragma unroll
    for (int j = 0; j < 4; ++j) p += qv[j] * krow[j * 32 + lane];
    for (int off = 16; off > 0; off >>= 1) p += __shfl_down(p, off);
    scores[l] = __shfl(p, 0) * SCALE_QK;
  }

  float mx = scores[0];
#pragma unroll
  for (int l = 1; l < L_LAYERS; ++l) mx = fmaxf(mx, scores[l]);
  float w[L_LAYERS], sum = 0.0f;
#pragma unroll
  for (int l = 0; l < L_LAYERS; ++l) { w[l] = __expf(scores[l] - mx); sum += w[l]; }
  const float inv = 1.0f / sum;

  float accv[16];
#pragma unroll
  for (int c = 0; c < 16; ++c) accv[c] = 0.0f;
#pragma unroll
  for (int l = 0; l < L_LAYERS; ++l) {
    const float* vrow = kqv + ((size_t)(l * SEQ + s)) * NCAT + QK + lane * 16;
    const float  wl   = w[l] * inv;
#pragma unroll
    for (int c = 0; c < 16; ++c) accv[c] += wl * vrow[c];
  }
#pragma unroll
  for (int c = 0; c < 16; ++c) agg[(size_t)s * HID + lane * 16 + c] = accv[c];

  if (s == SEQ - 1 && lane < L_LAYERS) attn_last[lane] = w[lane] * inv;
}

// ---------------------------------------------------------------------------
// hidden = relu(agg @ W1 + b1), emitted directly as bf16 for the logits GEMM.
// ---------------------------------------------------------------------------
__global__ void mlp1_kernel(const float* __restrict__ agg,
                            const float* __restrict__ W1,
                            const float* __restrict__ b1,
                            unsigned short* __restrict__ hidden_bf) {
  int idx = blockIdx.x * blockDim.x + threadIdx.x;   // 1024*256
  if (idx >= SEQ * (HID / 2)) return;
  int s = idx >> 8, n = idx & 255;
  float acc = b1[n];
  const float* arow = agg + (size_t)s * HID;
#pragma unroll 4
  for (int k = 0; k < HID; ++k) acc += arow[k] * W1[(size_t)k * (HID / 2) + n];
  hidden_bf[idx] = f2bf(fmaxf(acc, 0.0f));
}

// ---------------------------------------------------------------------------

extern "C" void kernel_launch(void* const* d_in, const int* in_sizes, int n_in,
                              void* d_out, int out_size, void* d_ws, size_t ws_size,
                              hipStream_t stream) {
  const float* h  = (const float*)d_in[0];
  const float* Wq = (const float*)d_in[1];
  const float* bq = (const float*)d_in[2];
  const float* Wk = (const float*)d_in[3];
  const float* bk = (const float*)d_in[4];
  const float* Wv = (const float*)d_in[5];
  const float* bv = (const float*)d_in[6];
  const float* W1 = (const float*)d_in[7];
  const float* b1 = (const float*)d_in[8];
  const float* W2 = (const float*)d_in[9];
  const float* b2 = (const float*)d_in[10];
  float* out = (float*)d_out;

  char* ws = (char*)d_ws;
  size_t off = 0;
  auto carve = [&](size_t bytes) {
    void* p = ws + off;
    off = (off + bytes + 255) & ~(size_t)255;
    return p;
  };
  unsigned short* wcat  = (unsigned short*)carve((size_t)(NCAT / 16) * 512 * 512 * 2);
  unsigned short* w2f   = (unsigned short*)carve((size_t)(VOCAB / 16) * 8 * 512 * 2);
  float* kqv            = (float*)carve((size_t)L_LAYERS * SEQ * NCAT * 4);
  float* q              = (float*)carve((size_t)SEQ * QK * 4);
  float* agg            = (float*)carve((size_t)SEQ * HID * 4);
  unsigned short* hidb  = (unsigned short*)carve((size_t)SEQ * (HID / 2) * 2);
  unsigned short* hbf   = (unsigned short*)carve((size_t)L_LAYERS * SEQ * SD * 2); // 384MB, last
  const bool bigws = (off <= ws_size);

  // 1) pack weights into bf16 WMMA B-fragments: [Wk | Wv | Wq] and W2
  pack_frags_kernel<<< (8  * 512 * 512) / 256, 256, 0, stream>>>(Wk, SD, QK,  wcat, 0,  512);
  pack_frags_kernel<<< (32 * 512 * 512) / 256, 256, 0, stream>>>(Wv, SD, HID, wcat, 8,  512);
  pack_frags_kernel<<< (8  * 512 * 512) / 256, 256, 0, stream>>>(Wq, SD, QK,  wcat, 40, 512);
  pack_frags_kernel<<< (2000 * 8 * 512) / 256, 256, 0, stream>>>(W2, 256, VOCAB, w2f, 0, 8);

  // 2) fused projection GEMM (12288x16384)@(16384x768)
  if (bigws) {
    f32_to_bf16_kernel<<<98304, 256, 0, stream>>>(h, hbf, 25165824LL);
    gemm_wmma_kernel<0, 0><<<dim3(NCAT / 128, (L_LAYERS * SEQ) / 128), 256, 0, stream>>>(
        hbf, SD, SD, wcat, 512, kqv, NCAT, bk, bv);
  } else {
    gemm_wmma_kernel<0, 1><<<dim3(NCAT / 128, (L_LAYERS * SEQ) / 128), 256, 0, stream>>>(
        h, SD, SD, wcat, 512, kqv, NCAT, bk, bv);
  }

  // 3) q = relu(mean_l qpart + bq)
  compute_q_kernel<<<(SEQ * QK) / 256, 256, 0, stream>>>(kqv, bq, q);

  // 4) attention over layers + weighted V aggregation (+ attn_last)
  attn_agg_kernel<<<SEQ, 32, 0, stream>>>(kqv, q, agg, out + (size_t)SEQ * VOCAB);

  // 5) hidden = relu(agg @ W1 + b1) -> bf16
  mlp1_kernel<<<(SEQ * (HID / 2)) / 256, 256, 0, stream>>>(agg, W1, b1, hidb);

  // 6) logits = hidden @ W2 + b2 -> d_out (async bf16 path)
  gemm_wmma_kernel<1, 0><<<dim3(VOCAB / 128, SEQ / 128), 256, 0, stream>>>(
      hidb, 256, 256, w2f, 8, out, VOCAB, b2, nullptr);
}
